// CSAI_18897856102810
// MI455X (gfx1250) — matrix operand order repaired and checked by reading, not compiled, not run on hardware
//
#include <hip/hip_runtime.h>

// ---------------------------------------------------------------------------
// CSAI (GRU-D style) scan for MI455X / gfx1250, wave32 + WMMA bf16 + TDM.
// B=4096, T=48, D=59 (pad 64), H=256, 2D=118 (pad 128), 3H=768.
// ---------------------------------------------------------------------------

typedef __attribute__((ext_vector_type(16))) __bf16 v16bf;
typedef __attribute__((ext_vector_type(8)))  float  v8f;
typedef unsigned int u32x4 __attribute__((ext_vector_type(4)));
typedef int          i32x4 __attribute__((ext_vector_type(4)));
typedef int          i32x8 __attribute__((ext_vector_type(8)));

#define NB 4096
#define NT 48
#define ND 59
#define NDP 64
#define NH 256
#define N2DP 128
#define N3H 768
#define NWG (NB/16)

#if defined(__has_builtin)
#if __has_builtin(__builtin_amdgcn_tensor_load_to_lds) && \
    __has_builtin(__builtin_amdgcn_s_wait_tensorcnt)
#define HAVE_TDM 1
#endif
#endif
#ifndef HAVE_TDM
#define HAVE_TDM 0
#endif

__device__ __forceinline__ unsigned short f2bf(float f) {
    union { float f; unsigned u; } v; v.f = f;
    unsigned r = v.u + 0x7FFFu + ((v.u >> 16) & 1u);
    return (unsigned short)(r >> 16);
}
__device__ __forceinline__ float bf2f(unsigned short s) {
    union { unsigned u; float f; } v; v.u = ((unsigned)s) << 16; return v.f;
}

// A-fragment: 16x32 bf16 tile (rows M=0..15, K window [k0,k0+32)), row-major LDS.
// lanes 0-15 row M=lane: e(0..15)->k0+e;
// lanes 16-31: same rows, +16 K offset.
__device__ __forceinline__ v16bf load_afrag(const unsigned short* base, int stride,
                                            int k0, int lane) {
    int m   = lane & 15;
    int off = (lane & 16) ? 16 : 0;
    const unsigned short* p = base + m * stride + k0 + off;
    union { unsigned u[8]; v16bf v; } r;
#pragma unroll
    for (int j = 0; j < 8; ++j) r.u[j] = *(const unsigned int*)(p + 2 * j);
    return r.v;
}

// B-fragment: 32x16 tile; weight row-major [N][K] so B[k][n]=W[n][k].
// Lane column n = nb+lane%16; lanes 16-31 take K offset +16; elements e -> k0+e.
__device__ __forceinline__ v16bf load_bfrag(const unsigned short* W, int stride,
                                            int nb, int k0, int lane) {
    int n = nb + (lane & 15);
    const unsigned short* p = W + n * stride + k0 + ((lane & 16) ? 16 : 0);
    union { unsigned u[8]; v16bf v; } r;
#pragma unroll
    for (int j = 0; j < 8; ++j) r.u[j] = *(const unsigned int*)(p + 2 * j);
    return r.v;
}

__device__ __forceinline__ v8f wmma_bf16(v16bf a, v16bf b, v8f c) {
    return __builtin_amdgcn_wmma_f32_16x16x32_bf16(false, a, false, b,
                                                   (short)0, c, false, false);
}

__device__ __forceinline__ float sigm(float v) { return 1.f / (1.f + __expf(-v)); }

// Tensor Data Mover: async DMA of a [16 x 59] f32 tile (row stride 48*59 elems)
// from global into contiguous LDS at lds_off.  D# per CDNA5 ISA.
// This toolchain (clang-23 / therock-10.0) uses the 6-arg builtin:
//   (u32x4 g0, i32x8 g1, i32x4 g2, i32x4 g3, i32x8 pad, i32 cpol)
__device__ __forceinline__ void tdm_load_tile(const float* gptr, unsigned lds_off) {
#if HAVE_TDM
    unsigned long long ga = (unsigned long long)gptr;
    u32x4 g0;
    g0.x = 1u;                                              // count=1 (valid user D#)
    g0.y = lds_off;                                         // lds_addr (bytes)
    g0.z = (unsigned)(ga & 0xFFFFFFFFu);                    // global_addr[31:0]
    g0.w = (unsigned)((ga >> 32) & 0x1FFFFFFu) | (2u << 30); // global_addr[56:32]|type=2
    i32x8 g1;
    g1[0] = (int)(2u << 16);        // workgroup_mask=0, data_size=2 (4 bytes)
    g1[1] = (int)(59u << 16);       // tensor_dim0 = 59
    g1[2] = (int)(16u << 16);       // tensor_dim1 = 16
    g1[3] = (int)(59u << 16);       // tile_dim0 = 59
    g1[4] = 16;                     // tile_dim1 = 16, tile_dim2 = 0
    g1[5] = NT * ND;                // tensor_dim0_stride = 2832 (lo32)
    g1[6] = 0;                      // stride hi, tensor_dim1_stride lo
    g1[7] = 0;
    i32x4 z4 = {0, 0, 0, 0};
    i32x8 z8 = {0, 0, 0, 0, 0, 0, 0, 0};
    __builtin_amdgcn_tensor_load_to_lds(g0, g1, z4, z4, z8, 0);
#else
    (void)gptr; (void)lds_off;
#endif
}

// ---------------------------------------------------------------------------
// Prep: convert f32 weights to padded bf16 layouts in workspace.
// ---------------------------------------------------------------------------
__global__ void csai_prep(const float* __restrict__ W_dh, const float* __restrict__ hist_W,
                          const float* __restrict__ feat_W, const float* __restrict__ wcomb_W,
                          const float* __restrict__ W_ih, const float* __restrict__ W_hh,
                          unsigned short* __restrict__ WdhB, unsigned short* __restrict__ histB,
                          unsigned short* __restrict__ featB, unsigned short* __restrict__ wcombB,
                          unsigned short* __restrict__ WihB, unsigned short* __restrict__ WhhB) {
    int i0 = blockIdx.x * blockDim.x + threadIdx.x;
    int st = gridDim.x * blockDim.x;
    for (int i = i0; i < NH * NDP; i += st) {                      // [256][64]
        int n = i >> 6, k = i & 63;
        WdhB[i] = f2bf(k < ND ? W_dh[n * ND + k] : 0.f);
    }
    for (int i = i0; i < NDP * NH; i += st) {                      // [64][256]
        int n = i >> 8, k = i & 255;
        histB[i] = f2bf(n < ND ? hist_W[n * NH + k] : 0.f);
    }
    for (int i = i0; i < NDP * NDP; i += st) {                     // [64][64] off-diag
        int n = i >> 6, k = i & 63;
        featB[i] = f2bf((n < ND && k < ND && n != k) ? feat_W[n * ND + k] : 0.f);
    }
    for (int i = i0; i < NDP * N2DP; i += st) {                    // [64][128]
        int n = i >> 7, kp = i & 127;
        float v = 0.f;
        if (n < ND) {
            if (kp < ND) v = wcomb_W[n * 118 + kp];
            else if (kp >= 64 && kp < 64 + ND) v = wcomb_W[n * 118 + ND + (kp - 64)];
        }
        wcombB[i] = f2bf(v);
    }
    for (int i = i0; i < N3H * N2DP; i += st) {                    // [768][128]
        int n = i >> 7, kp = i & 127;
        float v = 0.f;
        if (kp < ND) v = W_ih[n * 118 + kp];
        else if (kp >= 64 && kp < 64 + ND) v = W_ih[n * 118 + ND + (kp - 64)];
        WihB[i] = f2bf(v);
    }
    for (int i = i0; i < N3H * NH; i += st)                        // [768][256]
        WhhB[i] = f2bf(W_hh[i]);
}

// ---------------------------------------------------------------------------
// Persistent scan: 256 blocks x 256 threads (8 waves); block owns 16 batch rows.
// ---------------------------------------------------------------------------
__global__ __launch_bounds__(256) void csai_scan(
    const float* __restrict__ x, const float* __restrict__ mask,
    const float* __restrict__ deltas, const float* __restrict__ h0,
    const float* __restrict__ W_dx, const float* __restrict__ b_dx,
    const float* __restrict__ b_dh, const float* __restrict__ hist_b,
    const float* __restrict__ feat_b, const float* __restrict__ wcomb_b,
    const float* __restrict__ b_ih, const float* __restrict__ b_hh,
    const float* __restrict__ cls_W, const float* __restrict__ cls_b,
    const unsigned short* __restrict__ WdhB, const unsigned short* __restrict__ histB,
    const unsigned short* __restrict__ featB, const unsigned short* __restrict__ wcombB,
    const unsigned short* __restrict__ WihB, const unsigned short* __restrict__ WhhB,
    float* __restrict__ pNum, float* __restrict__ pDen,
    float* __restrict__ out_ximp, float* __restrict__ out_hid,
    float* __restrict__ out_y, float* __restrict__ out_score) {
    __shared__ float          stage_x[16 * ND];  // 3776 B  TDM-staged x_t
    __shared__ float          stage_m[16 * ND];  // 3776 B  TDM-staged mask_t
    __shared__ float          stage_d[16 * ND];  // 3776 B  TDM-staged deltas_t
    __shared__ float          hS[16][NH];        // 16 KB   current hidden (f32)
    __shared__ float          xhS[16][NDP];      // 4 KB    history estimate
    __shared__ unsigned short hB[16][NH];        // 8 KB    h as bf16 (WMMA A)
    __shared__ unsigned short rS[16][NH];        // 8 KB    GRU r gate (bf16)
    __shared__ unsigned short zS[16][NH];        // 8 KB    GRU z gate (bf16)
    __shared__ unsigned short dB[16][NDP];       // 2 KB    d_t bf16 -> beta
    __shared__ unsigned short xrB[16][NDP];      // 2 KB    x_r (A) -> xu
    __shared__ unsigned short catB[16][N2DP];    // 4 KB    concat input (WMMA A)
    __shared__ float          red[16];           // loss partials per wave

    const int tid  = threadIdx.x;
    const int w    = tid >> 5;
    const int lane = tid & 31;
    const int bid  = blockIdx.x;
    const int bm   = bid * 16;
    const int mhi  = (lane & 16) ? 8 : 0;
    const int nl   = lane & 15;
    const int gbase0 = bm * NT * ND;   // element index of (b=bm, t=0, d=0)

    for (int i = tid; i < 16 * NH; i += 256) {
        int r = i >> 8, k = i & 255;
        float v = h0[(bm + r) * NH + k];
        hS[r][k] = v; hB[r][k] = f2bf(v);
    }
#if HAVE_TDM
    if (w == 0) {   // prefetch tiles for t=0 (TDM ignores EXEC; one wave issues)
        tdm_load_tile(x      + gbase0, (unsigned)(unsigned long long)&stage_x[0]);
        tdm_load_tile(mask   + gbase0, (unsigned)(unsigned long long)&stage_m[0]);
        tdm_load_tile(deltas + gbase0, (unsigned)(unsigned long long)&stage_d[0]);
    }
#endif
    __syncthreads();

    for (int t = 0; t < NT; ++t) {
        // keep loop-variant memory ops inside the loop (no scratch spilling of
        // hoisted weight fragments)
        asm volatile("" ::: "memory");

        // ---- staging valid for step t ------------------------------------
#if HAVE_TDM
        if (w == 0) __builtin_amdgcn_s_wait_tensorcnt(0);
#else
        for (int i = tid; i < 16 * ND; i += 256) {
            int r = i / ND, c = i % ND;
            int gi = gbase0 + t * ND + r * (NT * ND) + c;
            stage_x[i] = x[gi]; stage_m[i] = mask[gi]; stage_d[i] = deltas[gi];
        }
#endif
        __syncthreads();

        // ---- A: d_t -> bf16 padded WMMA A operand ------------------------
        for (int i = tid; i < 16 * NDP; i += 256) {
            int r = i >> 6, c = i & 63;
            dB[r][c] = f2bf(c < ND ? stage_d[r * ND + c] : 0.f);
        }
        __syncthreads();

        // ---- B: gamma_h = exp(-relu(d@W_dh^T+b)); h *= gamma_h ----------
        for (int s = 0; s < 2; ++s) {
            int nb = (w * 2 + s) * 16;
            v8f acc = {};
            for (int k0 = 0; k0 < NDP; k0 += 32)
                acc = wmma_bf16(load_afrag(&dB[0][0], NDP, k0, lane),
                                load_bfrag(WdhB, NDP, nb, k0, lane), acc);
            int n = nb + nl;
            float bias = b_dh[n];
#pragma unroll
            for (int i = 0; i < 8; ++i) {
                int m = i + mhi;
                float g = __expf(-fmaxf(acc[i] + bias, 0.f));
                float hv = hS[m][n] * g;
                hS[m][n] = hv; hB[m][n] = f2bf(hv);
            }
        }
        __syncthreads();

        // ---- C: x_h = h @ hist_W^T + hist_b (waves 0-3) ------------------
        if (w < 4) {
            int nb = w * 16;
            v8f acc = {};
            for (int k0 = 0; k0 < NH; k0 += 32)
                acc = wmma_bf16(load_afrag(&hB[0][0], NH, k0, lane),
                                load_bfrag(histB, NH, nb, k0, lane), acc);
            int n = nb + nl;
            float bias = (n < ND) ? hist_b[n] : 0.f;
#pragma unroll
            for (int i = 0; i < 8; ++i)
                xhS[i + mhi][n] = (n < ND) ? (acc[i] + bias) : 0.f;
        }
        __syncthreads();

        // ---- D: x_r, gamma_x, concat(gamma_x, m) -------------------------
        for (int i = tid; i < 16 * NDP; i += 256) {
            int r = i >> 6, c = i & 63;
            float m = 0.f, xv = 0.f, gx = 0.f;
            if (c < ND) {
                m  = stage_m[r * ND + c];
                xv = stage_x[r * ND + c];
                float dv = stage_d[r * ND + c];
                gx = __expf(-fmaxf(dv * W_dx[c * (ND + 1)] + b_dx[c], 0.f));
            }
            float xr = m * xv + (1.f - m) * xhS[r][c];
            xrB[r][c] = f2bf((c < ND) ? xr : 0.f);
            catB[r][c]      = f2bf(gx);
            catB[r][64 + c] = f2bf(m);
        }
        __syncthreads();

        // ---- E/F: xu (waves 0-3) | beta (waves 4-7); barrier-split so the
        //      results can reuse xrB / dB in place -------------------------
        v8f accEF = {};
        if (w < 4) {
            int nb = w * 16;
            for (int k0 = 0; k0 < NDP; k0 += 32)
                accEF = wmma_bf16(load_afrag(&xrB[0][0], NDP, k0, lane),
                                  load_bfrag(featB, NDP, nb, k0, lane), accEF);
        } else {
            int nb = (w - 4) * 16;
            for (int k0 = 0; k0 < N2DP; k0 += 32)
                accEF = wmma_bf16(load_afrag(&catB[0][0], N2DP, k0, lane),
                                  load_bfrag(wcombB, N2DP, nb, k0, lane), accEF);
        }
        __syncthreads();
        if (w < 4) {                       // xu -> xrB
            int n = w * 16 + nl;
            float bias = (n < ND) ? feat_b[n] : 0.f;
#pragma unroll
            for (int i = 0; i < 8; ++i)
                xrB[i + mhi][n] = f2bf((n < ND) ? (accEF[i] + bias) : 0.f);
        } else {                           // beta -> dB
            int n = (w - 4) * 16 + nl;
            float bias = (n < ND) ? wcomb_b[n] : 0.f;
#pragma unroll
            for (int i = 0; i < 8; ++i)
                dB[i + mhi][n] = f2bf(sigm(accEF[i] + bias));
        }
        __syncthreads();

        // ---- G: x_comb, loss, x_imp, concat(x_imp, m) --------------------
        float lnum = 0.f, lden = 0.f;
        for (int i = tid; i < 16 * NDP; i += 256) {
            int r = i >> 6, c = i & 63;
            if (c < ND) {
                float beta  = bf2f(dB[r][c]);
                float xu    = bf2f(xrB[r][c]);
                float xcomb = beta * xu + (1.f - beta) * xhS[r][c];
                float xv = stage_x[r * ND + c];
                float m  = stage_m[r * ND + c];
                float xi = m * xv + (1.f - m) * xcomb;
                lnum += fabsf(xv - xcomb) * m;
                lden += m;
                out_ximp[((bm + r) * NT + t) * ND + c] = xi;
                catB[r][c]      = f2bf(xi);
                catB[r][64 + c] = f2bf(m);
            } else {
                catB[r][c] = 0; catB[r][64 + c] = 0;
            }
        }
#pragma unroll
        for (int o = 16; o > 0; o >>= 1) {
            lnum += __shfl_xor(lnum, o, 32);
            lden += __shfl_xor(lden, o, 32);
        }
        if (lane == 0) { red[w] = lnum; red[8 + w] = lden; }
        __syncthreads();
        if (tid == 0) {
            float n = 0.f, d = 0.f;
            for (int i = 0; i < 8; ++i) { n += red[i]; d += red[8 + i]; }
            pNum[t * NWG + bid] = n; pDen[t * NWG + bid] = d;
        }
#if HAVE_TDM
        // staging fully consumed -> overlap next step's tile DMA with H/I
        if (w == 0 && t + 1 < NT) {
            int gb = gbase0 + (t + 1) * ND;
            tdm_load_tile(x      + gb, (unsigned)(unsigned long long)&stage_x[0]);
            tdm_load_tile(mask   + gb, (unsigned)(unsigned long long)&stage_m[0]);
            tdm_load_tile(deltas + gb, (unsigned)(unsigned long long)&stage_d[0]);
        }
#endif

        // ---- H: GRU r,z gates: sigmoid(cat@Wih + h@Whh + biases) ---------
        for (int s = 0; s < 4; ++s) {
            int nb = (w * 4 + s) * 16;                 // gate cols [0,512)
            v8f acc = {};
            for (int k0 = 0; k0 < N2DP; k0 += 32)
                acc = wmma_bf16(load_afrag(&catB[0][0], N2DP, k0, lane),
                                load_bfrag(WihB, N2DP, nb, k0, lane), acc);
            for (int k0 = 0; k0 < NH; k0 += 32)
                acc = wmma_bf16(load_afrag(&hB[0][0], NH, k0, lane),
                                load_bfrag(WhhB, NH, nb, k0, lane), acc);
            int n = nb + nl;
            float bias = b_ih[n] + b_hh[n];
#pragma unroll
            for (int i = 0; i < 8; ++i) {
                float sg = sigm(acc[i] + bias);
                if (n < NH) rS[i + mhi][n]      = f2bf(sg);
                else        zS[i + mhi][n - NH] = f2bf(sg);
            }
        }
        __syncthreads();

        // ---- I: n gate (gi, gh kept separate) + state update -------------
        v8f gi0 = {}, gh0 = {}, gi1 = {}, gh1 = {};
        for (int s = 0; s < 2; ++s) {
            int nb = (w * 2 + s) * 16;
            v8f gi = {}, gh = {};
            for (int k0 = 0; k0 < N2DP; k0 += 32)
                gi = wmma_bf16(load_afrag(&catB[0][0], N2DP, k0, lane),
                               load_bfrag(WihB, N2DP, 512 + nb, k0, lane), gi);
            for (int k0 = 0; k0 < NH; k0 += 32)
                gh = wmma_bf16(load_afrag(&hB[0][0], NH, k0, lane),
                               load_bfrag(WhhB, NH, 512 + nb, k0, lane), gh);
            if (s == 0) { gi0 = gi; gh0 = gh; } else { gi1 = gi; gh1 = gh; }
        }
        __syncthreads();   // all hB reads complete before overwrite
        for (int s = 0; s < 2; ++s) {
            int n  = (w * 2 + s) * 16 + nl;
            v8f gi = s ? gi1 : gi0;
            v8f gh = s ? gh1 : gh0;
            float bi = b_ih[512 + n], bh = b_hh[512 + n];
#pragma unroll
            for (int i = 0; i < 8; ++i) {
                int m = i + mhi;
                float r  = bf2f(rS[m][n]);
                float z  = bf2f(zS[m][n]);
                float nn = tanhf((gi[i] + bi) + r * (gh[i] + bh));
                float hv = (1.f - z) * nn + z * hS[m][n];
                hS[m][n] = hv; hB[m][n] = f2bf(hv);
                out_hid[((bm + m) * NT + t) * NH + n] = hv;
            }
        }
        __syncthreads();
    }

    // ---- classification head on final h ---------------------------------
    for (int s = 0; s < 2; ++s) {
        int r = w + 8 * s;
        float p = 0.f;
        for (int k = lane; k < NH; k += 32) p += hS[r][k] * cls_W[k];
#pragma unroll
        for (int o = 16; o > 0; o >>= 1) p += __shfl_xor(p, o, 32);
        if (lane == 0) {
            float yo = p + cls_b[0];
            out_y[bm + r] = yo;
            out_score[bm + r] = sigm(yo);
        }
    }
}

// ---------------------------------------------------------------------------
// decay_factor = 0.5*(1 - tanh(sign(dd)*|dd@wobs_W^T + b|)), dd = deltas - med
// ---------------------------------------------------------------------------
__global__ void csai_decay(const float* __restrict__ deltas, const float* __restrict__ medians,
                           const float* __restrict__ wobsW, const float* __restrict__ wobsb,
                           float* __restrict__ out) {
    int i = blockIdx.x * blockDim.x + threadIdx.x;
    if (i >= NB * NT * ND) return;
    int e = i % ND, bt = i / ND;
    const float* drow = deltas + bt * ND;
    float acc = 0.f;
    for (int d = 0; d < ND; ++d)
        acc += (drow[d] - medians[d]) * wobsW[e * ND + d];
    float wd  = acc + wobsb[e];
    float dde = drow[e] - medians[e];
    float s   = (dde > 0.f) ? 1.f : ((dde < 0.f) ? -1.f : 0.f);
    out[i] = 0.5f * (1.f - tanhf(s * fabsf(wd)));
}

// ---------------------------------------------------------------------------
// Deterministic loss reduction: x_loss = sum_t sum_wg(num)/(sum_wg(den)+1e-5)
// ---------------------------------------------------------------------------
__global__ void csai_finalize(const float* __restrict__ pNum, const float* __restrict__ pDen,
                              float* __restrict__ out_loss) {
    __shared__ float lt[NT];
    int t = threadIdx.x;
    if (t < NT) {
        float n = 0.f, d = 0.f;
        for (int wg = 0; wg < NWG; ++wg) { n += pNum[t * NWG + wg]; d += pDen[t * NWG + wg]; }
        lt[t] = n / (d + 1e-5f);
    }
    __syncthreads();
    if (t == 0) {
        float s = 0.f;
        for (int i = 0; i < NT; ++i) s += lt[i];
        *out_loss = s;
    }
}

extern "C" void kernel_launch(void* const* d_in, const int* in_sizes, int n_in,
                              void* d_out, int out_size, void* d_ws, size_t ws_size,
                              hipStream_t stream) {
    const float* x       = (const float*)d_in[0];
    const float* mask    = (const float*)d_in[1];
    const float* deltas  = (const float*)d_in[2];
    // d_in[3] = last_obs (unused by the reference body)
    const float* h0      = (const float*)d_in[4];
    const float* medians = (const float*)d_in[5];
    const float* W_dh    = (const float*)d_in[6];
    const float* b_dh    = (const float*)d_in[7];
    const float* W_dx    = (const float*)d_in[8];
    const float* b_dx    = (const float*)d_in[9];
    const float* hist_W  = (const float*)d_in[10];
    const float* hist_b  = (const float*)d_in[11];
    const float* feat_W  = (const float*)d_in[12];
    const float* feat_b  = (const float*)d_in[13];
    const float* wcomb_W = (const float*)d_in[14];
    const float* wcomb_b = (const float*)d_in[15];
    const float* wobs_W  = (const float*)d_in[16];
    const float* wobs_b  = (const float*)d_in[17];
    const float* W_ih    = (const float*)d_in[18];
    const float* W_hh    = (const float*)d_in[19];
    const float* b_ih    = (const float*)d_in[20];
    const float* b_hh    = (const float*)d_in[21];
    const float* cls_W   = (const float*)d_in[22];
    const float* cls_b   = (const float*)d_in[23];

    char* ws = (char*)d_ws;
    unsigned short* WdhB   = (unsigned short*)ws; ws += NH * NDP * 2;
    unsigned short* histB  = (unsigned short*)ws; ws += NDP * NH * 2;
    unsigned short* featB  = (unsigned short*)ws; ws += NDP * NDP * 2;
    unsigned short* wcombB = (unsigned short*)ws; ws += NDP * N2DP * 2;
    unsigned short* WihB   = (unsigned short*)ws; ws += N3H * N2DP * 2;
    unsigned short* WhhB   = (unsigned short*)ws; ws += N3H * NH * 2;
    float* pNum = (float*)ws; ws += NT * NWG * 4;
    float* pDen = (float*)ws; ws += NT * NWG * 4;

    float* out       = (float*)d_out;
    float* out_ximp  = out;                    // [B,T,D]  11599872
    float* out_loss  = out + 11599872;         // scalar
    float* out_hid   = out + 11599873;         // [B,T,H]  50331648
    float* out_y     = out + 61931521;         // [B,1]
    float* out_score = out + 61935617;         // [B,1]
    float* out_decay = out + 61939713;         // [B,T,D]

    csai_prep<<<256, 256, 0, stream>>>(W_dh, hist_W, feat_W, wcomb_W, W_ih, W_hh,
                                       WdhB, histB, featB, wcombB, WihB, WhhB);
    csai_scan<<<NWG, 256, 0, stream>>>(x, mask, deltas, h0, W_dx, b_dx, b_dh, hist_b,
                                       feat_b, wcomb_b, b_ih, b_hh, cls_W, cls_b,
                                       WdhB, histB, featB, wcombB, WihB, WhhB,
                                       pNum, pDen, out_ximp, out_hid, out_y, out_score);
    csai_decay<<<(NB * NT * ND + 255) / 256, 256, 0, stream>>>(deltas, medians, wobs_W,
                                                               wobs_b, out_decay);
    csai_finalize<<<1, 64, 0, stream>>>(pNum, pDen, out_loss);
}
